// GAT_44169443672086
// MI455X (gfx1250) — matrix operand order repaired and checked by reading, not compiled, not run on hardware
//
#include <hip/hip_runtime.h>
#include <hip/hip_bf16.h>

// ---------------------------------------------------------------------------
// GAT forward for MI455X (gfx1250, wave32, WMMA bf16).
//   N=4096 nodes, F=1024 feats, H=8 heads, d=64 hidden, C=32 classes.
// GEMM + flash-attention work rides v_wmma_f32_16x16x32_bf16; the head GEMM
// stages its shared W tile through LDS with the gfx1250 async global->LDS
// DMA path (ASYNCcnt / s_wait_asynccnt), double buffered. adj (64MB) is
// re-read per head out of the 192MB L2. Softmax uses a monotonicity-derived
// row upper-bound shift so the flash pass needs no online rescaling.
// ---------------------------------------------------------------------------

#define NN     4096
#define NF     1024
#define NHID   64
#define NHEAD  8
#define NCLASS 32
#define NCAT   (NHEAD*NHID)   // 512
#define ALPHA  0.2f

typedef __attribute__((ext_vector_type(16))) __bf16 v16bf;
typedef __attribute__((ext_vector_type(8)))  __bf16 v8bf;
typedef __attribute__((ext_vector_type(8)))  float  v8f;

__device__ __forceinline__ float lrelu(float x){ return x > 0.f ? x : ALPHA * x; }

// A fragment: 16x32 bf16 from row-major A (stride lda), tile at (row0, k0).
// Contiguous-k layout used here: lanes 0-15 -> M=lane, K = k0+0..15;
//             lanes 16-31 -> M=lane-16, K = k0+16..31.
__device__ __forceinline__ v16bf load_a_frag(const __bf16* A, int lda, int row0, int k0, int lane){
    int row  = row0 + (lane & 15);
    int koff = (lane >> 4) << 4;                 // 0 or 16
    const __bf16* p = A + (size_t)row * lda + k0 + koff;
    v8bf lo = *(const v8bf*)(p);
    v8bf hi = *(const v8bf*)(p + 8);
    v16bf a;
#pragma unroll
    for (int t = 0; t < 8; ++t){ a[t] = lo[t]; a[8+t] = hi[t]; }
    return a;
}

// B fragment: 32x16 bf16 from N-major (pre-transposed) Bt[n][k] (stride ldk).
// Contiguous-k layout used here: lanes 0-15 -> N=lane, K=k0+0..15; lanes 16-31 -> N=lane-16, K=k0+16..31.
__device__ __forceinline__ v16bf load_b_frag(const __bf16* Bt, int ldk, int n0, int k0, int lane){
    int n    = n0 + (lane & 15);
    int koff = (lane >> 4) << 4;                 // 0 or 16
    return *(const v16bf*)(Bt + (size_t)n * ldk + k0 + koff);
}

__device__ __forceinline__ v8f wmma_bf16(v16bf a, v16bf b, v8f c){
    return __builtin_amdgcn_wmma_f32_16x16x32_bf16(false, a, false, b, (short)0, c, false, false);
}

// gfx1250 async global->LDS DMA (tracked by ASYNCcnt). lds_off is the 32-bit
// LDS byte offset (low 32 bits of the flat shared-aperture address).
__device__ __forceinline__ void async_b128(unsigned lds_off, const void* g){
    asm volatile("global_load_async_to_lds_b128 %0, %1, off"
                 :: "v"(lds_off), "v"((unsigned long long)(uintptr_t)g) : "memory");
}
__device__ __forceinline__ void wait_async(int n){
    if (n == 0) asm volatile("s_wait_asynccnt 0x0" ::: "memory");
    else        asm volatile("s_wait_asynccnt 0x1" ::: "memory");
}

// ---------------------------------------------------------------- casts ----
__global__ void k_cast_x(const float* __restrict__ src, __bf16* __restrict__ dst, int n){
    int i = blockIdx.x * 256 + threadIdx.x;
    if (i < n) dst[i] = (__bf16)src[i];
}

// W_heads [H][K=1024][O=64] -> WtT [H][O][K] bf16
__global__ void k_transpose_Wheads(const float* __restrict__ W, __bf16* __restrict__ WtT){
    int h = blockIdx.y, o = blockIdx.x;
    for (int k = threadIdx.x; k < NF; k += 256)
        WtT[((size_t)h*NHID + o)*NF + k] = (__bf16)W[((size_t)h*NF + k)*NHID + o];
}

// W_out [512][32] -> WoutT [32][512] bf16
__global__ void k_transpose_Wout(const float* __restrict__ W, __bf16* __restrict__ WoutT){
    int c = blockIdx.x;
    for (int k = threadIdx.x; k < NCAT; k += 256)
        WoutT[(size_t)c*NCAT + k] = (__bf16)W[(size_t)k*NCLASS + c];
}

// ------------------------------------------------- head feature GEMM -------
// hbT[h][o][n] = bf16( (x @ W_h)[n][o] + b[h][o] )   (stored N-major for reuse as B)
// The 64x32 W tile (shared by all 8 waves) is staged via async global->LDS
// DMA, double-buffered; B fragments come from LDS (conflict-free 80B pitch).
// k-loop unrolled x2 so the LDS buffer index is compile-time per copy.
#define WPITCH 80   // bytes per staged W row (64B data + 16B pad, 16B aligned)
__global__ __launch_bounds__(256) void k_head_gemm(const __bf16* __restrict__ xb,
                                                   const __bf16* __restrict__ WtT,
                                                   const float*  __restrict__ b_heads,
                                                   __bf16* __restrict__ hbT){
    int head = blockIdx.y;
    int tid  = threadIdx.x;
    int wave = tid >> 5, lane = tid & 31;
    int row0 = (blockIdx.x * 8 + wave) * 16;
    const __bf16* Bt = WtT + (size_t)head * NHID * NF;

    __shared__ __align__(16) char sW[2][64 * WPITCH];

    // async staging role: thread t covers 16B chunk `sub` of W row `o`.
    int o_ld  = tid >> 2;                 // 0..63
    int sub   = tid & 3;                  // 0..3
    const __bf16* gld = Bt + (size_t)o_ld * NF + sub * 8;
    unsigned ldsA = (unsigned)(uintptr_t)&sW[0][o_ld * WPITCH + sub * 16];
    unsigned ldsB = (unsigned)(uintptr_t)&sW[1][o_ld * WPITCH + sub * 16];

    async_b128(ldsA, gld);                // stage k-step 0 into buffer 0

    v8f c[4];
#pragma unroll
    for (int nt = 0; nt < 4; ++nt){ v8f z = {0.f,0.f,0.f,0.f,0.f,0.f,0.f,0.f}; c[nt] = z; }

    int koffB = (lane >> 4) << 5;         // byte offset in W row: 0 or 32
    v16bf a_cur = load_a_frag(xb, NF, row0, 0, lane);

    const int steps = NF / 32;
#pragma unroll 2
    for (int step = 0; step < steps; ++step){
        if (step + 1 < steps)
            async_b128((step & 1) ? ldsA : ldsB, gld + (step + 1) * 32);
        wait_async(step + 1 < steps ? 1 : 0);   // stage `step` landed
        __syncthreads();

        v16bf a_next = a_cur;
        if (step + 1 < steps)
            a_next = load_a_frag(xb, NF, row0, (step + 1) * 32, lane);

        // batch all 8 ds_load_b128 before the 4 WMMAs -> single DS wait
        v16bf b[4];
#pragma unroll
        for (int nt = 0; nt < 4; ++nt){
            int o = nt*16 + (lane & 15);
            b[nt] = *(const v16bf*)&sW[step & 1][o * WPITCH + koffB];
        }
#pragma unroll
        for (int nt = 0; nt < 4; ++nt)
            c[nt] = wmma_bf16(a_cur, b[nt], c[nt]);

        a_cur = a_next;
        __syncthreads();                  // computes done before buffer reuse
    }

    int m0 = (lane >> 4) << 3;            // C layout: lanes 0-15 rows 0-7, 16-31 rows 8-15
#pragma unroll
    for (int nt = 0; nt < 4; ++nt){
        int o = nt*16 + (lane & 15);
        float bias = b_heads[head*NHID + o];
        __bf16* dst = hbT + ((size_t)head*NHID + o) * NN;
#pragma unroll
        for (int r = 0; r < 8; ++r)
            dst[row0 + m0 + r] = (__bf16)(c[nt][r] + bias);
    }
}

// ------------------------------------- per-head scores + global dst-max ----
__global__ __launch_bounds__(256) void k_scores_head(const __bf16* __restrict__ hbT,
                                                     const float* __restrict__ a_heads,
                                                     float* __restrict__ ssrc,
                                                     float* __restrict__ sdst,
                                                     float* __restrict__ gmax){
    int head = blockIdx.x;
    __shared__ float smax[256];
    float mymax = -3.0e38f;
    for (int n = threadIdx.x; n < NN; n += 256){
        float ss = 0.f, sd = 0.f;
#pragma unroll 8
        for (int o = 0; o < NHID; ++o){
            float v = (float)hbT[((size_t)head*NHID + o)*NN + n];     // coalesced over n
            ss += v * a_heads[head*2*NHID + o];
            sd += v * a_heads[head*2*NHID + NHID + o];
        }
        ssrc[head*NN + n] = ss;
        sdst[head*NN + n] = sd;
        mymax = fmaxf(mymax, sd);
    }
    smax[threadIdx.x] = mymax; __syncthreads();
    for (int s = 128; s > 0; s >>= 1){
        if (threadIdx.x < s) smax[threadIdx.x] = fmaxf(smax[threadIdx.x], smax[threadIdx.x + s]);
        __syncthreads();
    }
    if (threadIdx.x == 0) gmax[head] = smax[0];
}

// ----------------------------------------- fused flash attention (heads) ---
// block = (rowTile, head); 4 waves each own a 1024-wide j slice; P tiles are
// built in registers in A-fragment layout and fed straight into WMMA. B-frag
// loads are hoisted so the exp() VALU work overlaps them; unroll-2 lets the
// scheduler hoist iteration s+1 loads over iteration s WMMAs.
__global__ __launch_bounds__(128) void k_attention_heads(const int* __restrict__ adj,
                                                         const __bf16* __restrict__ hbT,
                                                         const float* __restrict__ ssrc,
                                                         const float* __restrict__ sdst,
                                                         const float* __restrict__ gmax,
                                                         __bf16* __restrict__ xcat){
    int head = blockIdx.y, rowTile = blockIdx.x;
    int wave = threadIdx.x >> 5, lane = threadIdx.x & 31;
    int row  = lane & 15;
    int i    = rowTile*16 + row;

    float ssrc_i = ssrc[head*NN + i];
    float shift  = lrelu(ssrc_i + gmax[head]);    // row-wise upper bound on all scores
    const int*    adjRow = adj + (size_t)i * NN;
    const float*  sd     = sdst + head*NN;
    const __bf16* Bt     = hbT + (size_t)head*NHID*NN;
    int koff = (lane >> 4) << 4;                  // A-frag K offset: 0 or 16
    int jwb  = wave * (NN/4);

    v8f acc[4];
#pragma unroll
    for (int nt = 0; nt < 4; ++nt){ v8f z = {0.f,0.f,0.f,0.f,0.f,0.f,0.f,0.f}; acc[nt] = z; }
    float psum = 0.f;

#pragma unroll 2
    for (int step = 0; step < (NN/4)/32; ++step){
        int js = jwb + step*32;
        // B fragments first: their latency overlaps the exp() work below.
        v16bf bfr[4];
#pragma unroll
        for (int nt = 0; nt < 4; ++nt) bfr[nt] = load_b_frag(Bt, NN, nt*16, js, lane);
        __builtin_prefetch(adjRow + js + 64, 0, 1);   // global_prefetch two steps ahead
        v16bf p;
        float lsum = 0.f;
#pragma unroll
        for (int hf = 0; hf < 2; ++hf){
            int jb = js + koff + hf*8;
            int4   m0 = *(const int4*)  (adjRow + jb);
            int4   m1 = *(const int4*)  (adjRow + jb + 4);
            float4 d0 = *(const float4*)(sd + jb);
            float4 d1 = *(const float4*)(sd + jb + 4);
            int   ms[8]  = {m0.x,m0.y,m0.z,m0.w,m1.x,m1.y,m1.z,m1.w};
            float dsv[8] = {d0.x,d0.y,d0.z,d0.w,d1.x,d1.y,d1.z,d1.w};
#pragma unroll
            for (int t = 0; t < 8; ++t){
                float e  = lrelu(ssrc_i + dsv[t]);
                float pv = (ms[t] > 0) ? __expf(e - shift) : 0.f;   // in (0,1]
                lsum += pv;
                p[hf*8 + t] = (__bf16)pv;
            }
        }
        psum += lsum;
#pragma unroll
        for (int nt = 0; nt < 4; ++nt)
            acc[nt] = wmma_bf16(p, bfr[nt], acc[nt]);
    }

    // combine the 4 j-slices (pure sums thanks to the fixed shift), normalize,
    // ELU, and scatter into the concat layout x_cat[n][head*64 + o] (bf16).
    __shared__ float sAcc[4][16][NHID];
    __shared__ float sSum[4][32];
    __shared__ float sRow[16];
    int m0r = (lane >> 4) << 3;
#pragma unroll
    for (int nt = 0; nt < 4; ++nt){
        int col = nt*16 + row;
#pragma unroll
        for (int r = 0; r < 8; ++r) sAcc[wave][m0r + r][col] = acc[nt][r];
    }
    sSum[wave][lane] = psum;
    __syncthreads();
    if (threadIdx.x < 16){
        float s = 0.f;
#pragma unroll
        for (int w = 0; w < 4; ++w) s += sSum[w][threadIdx.x] + sSum[w][threadIdx.x + 16];
        sRow[threadIdx.x] = fmaxf(s, 1e-30f);
    }
    __syncthreads();
    for (int idx = threadIdx.x; idx < 16*NHID; idx += 128){
        int r = idx >> 6, col = idx & 63;
        float v = (sAcc[0][r][col] + sAcc[1][r][col] + sAcc[2][r][col] + sAcc[3][r][col]) / sRow[r];
        v = v > 0.f ? v : (__expf(v) - 1.f);      // ELU
        xcat[(size_t)(rowTile*16 + r)*NCAT + head*NHID + col] = (__bf16)v;
    }
}

// ------------------------------------------------- output GEMM -------------
// houtT[c][n] = bf16( (x_cat @ W_out)[n][c] + b_out[c] )
__global__ __launch_bounds__(256) void k_out_gemm(const __bf16* __restrict__ xcat,
                                                  const __bf16* __restrict__ WoutT,
                                                  const float*  __restrict__ b_out,
                                                  __bf16* __restrict__ houtT){
    int wave = threadIdx.x >> 5, lane = threadIdx.x & 31;
    int row0 = (blockIdx.x * 8 + wave) * 16;
    v8f c[2];
#pragma unroll
    for (int nt = 0; nt < 2; ++nt){ v8f z = {0.f,0.f,0.f,0.f,0.f,0.f,0.f,0.f}; c[nt] = z; }
#pragma unroll 2
    for (int k0 = 0; k0 < NCAT; k0 += 32){
        v16bf a  = load_a_frag(xcat, NCAT, row0, k0, lane);
        v16bf b0 = load_b_frag(WoutT, NCAT, 0,  k0, lane);
        v16bf b1 = load_b_frag(WoutT, NCAT, 16, k0, lane);
        c[0] = wmma_bf16(a, b0, c[0]);
        c[1] = wmma_bf16(a, b1, c[1]);
    }
    int m0 = (lane >> 4) << 3;
#pragma unroll
    for (int nt = 0; nt < 2; ++nt){
        int cl = nt*16 + (lane & 15);
        float bias = b_out[cl];
#pragma unroll
        for (int r = 0; r < 8; ++r)
            houtT[(size_t)cl*NN + row0 + m0 + r] = (__bf16)(c[nt][r] + bias);
    }
}

// ------------------------------------ output-layer scores + dst-max --------
__global__ __launch_bounds__(256) void k_scores_out(const __bf16* __restrict__ houtT,
                                                    const float* __restrict__ a_out,
                                                    float* __restrict__ s2src,
                                                    float* __restrict__ s2dst,
                                                    float* __restrict__ gmax2){
    __shared__ float smax[256];
    float mymax = -3.0e38f;
    for (int n = threadIdx.x; n < NN; n += 256){
        float ss = 0.f, sdv = 0.f;
#pragma unroll
        for (int o = 0; o < NCLASS; ++o){
            float v = (float)houtT[(size_t)o*NN + n];
            ss  += v * a_out[o];
            sdv += v * a_out[NCLASS + o];
        }
        s2src[n] = ss; s2dst[n] = sdv;
        mymax = fmaxf(mymax, sdv);
    }
    smax[threadIdx.x] = mymax; __syncthreads();
    for (int s = 128; s > 0; s >>= 1){
        if (threadIdx.x < s) smax[threadIdx.x] = fmaxf(smax[threadIdx.x], smax[threadIdx.x + s]);
        __syncthreads();
    }
    if (threadIdx.x == 0) gmax2[0] = smax[0];
}

// --------------------------- fused output attention + log_softmax ----------
__global__ __launch_bounds__(128) void k_attention_out(const int* __restrict__ adj,
                                                       const __bf16* __restrict__ houtT,
                                                       const float* __restrict__ s2src,
                                                       const float* __restrict__ s2dst,
                                                       const float* __restrict__ gmax2,
                                                       float* __restrict__ out){
    int rowTile = blockIdx.x;
    int wave = threadIdx.x >> 5, lane = threadIdx.x & 31;
    int row  = lane & 15;
    int i    = rowTile*16 + row;

    float ssrc_i = s2src[i];
    float shift  = lrelu(ssrc_i + gmax2[0]);
    const int* adjRow = adj + (size_t)i * NN;
    int koff = (lane >> 4) << 4;
    int jwb  = wave * (NN/4);

    v8f acc[2];
#pragma unroll
    for (int nt = 0; nt < 2; ++nt){ v8f z = {0.f,0.f,0.f,0.f,0.f,0.f,0.f,0.f}; acc[nt] = z; }
    float psum = 0.f;

#pragma unroll 2
    for (int step = 0; step < (NN/4)/32; ++step){
        int js = jwb + step*32;
        v16bf b0 = load_b_frag(houtT, NN, 0,  js, lane);
        v16bf b1 = load_b_frag(houtT, NN, 16, js, lane);
        __builtin_prefetch(adjRow + js + 64, 0, 1);
        v16bf p;
        float lsum = 0.f;
#pragma unroll
        for (int hf = 0; hf < 2; ++hf){
            int jb = js + koff + hf*8;
            int4   m0 = *(const int4*)  (adjRow + jb);
            int4   m1 = *(const int4*)  (adjRow + jb + 4);
            float4 d0 = *(const float4*)(s2dst + jb);
            float4 d1 = *(const float4*)(s2dst + jb + 4);
            int   ms[8]  = {m0.x,m0.y,m0.z,m0.w,m1.x,m1.y,m1.z,m1.w};
            float dsv[8] = {d0.x,d0.y,d0.z,d0.w,d1.x,d1.y,d1.z,d1.w};
#pragma unroll
            for (int t = 0; t < 8; ++t){
                float e  = lrelu(ssrc_i + dsv[t]);
                float pv = (ms[t] > 0) ? __expf(e - shift) : 0.f;
                lsum += pv;
                p[hf*8 + t] = (__bf16)pv;
            }
        }
        psum += lsum;
        acc[0] = wmma_bf16(p, b0, acc[0]);
        acc[1] = wmma_bf16(p, b1, acc[1]);
    }

    __shared__ float sAcc[4][16][NCLASS];
    __shared__ float sSum[4][32];
    __shared__ float sRow[16];
    __shared__ float sOut[16][NCLASS];
    int m0r = (lane >> 4) << 3;
#pragma unroll
    for (int nt = 0; nt < 2; ++nt){
        int col = nt*16 + row;
#pragma unroll
        for (int r = 0; r < 8; ++r) sAcc[wave][m0r + r][col] = acc[nt][r];
    }
    sSum[wave][lane] = psum;
    __syncthreads();
    if (threadIdx.x < 16){
        float s = 0.f;
#pragma unroll
        for (int w = 0; w < 4; ++w) s += sSum[w][threadIdx.x] + sSum[w][threadIdx.x + 16];
        sRow[threadIdx.x] = fmaxf(s, 1e-30f);
    }
    __syncthreads();
    for (int idx = threadIdx.x; idx < 16*NCLASS; idx += 128){
        int r = idx >> 5, cc = idx & 31;
        sOut[r][cc] = (sAcc[0][r][cc] + sAcc[1][r][cc] + sAcc[2][r][cc] + sAcc[3][r][cc]) / sRow[r];
    }
    __syncthreads();
    if (threadIdx.x < 16){
        int r = threadIdx.x;
        float m = -3.0e38f;
#pragma unroll
        for (int cc = 0; cc < NCLASS; ++cc) m = fmaxf(m, sOut[r][cc]);
        float l = 0.f;
#pragma unroll
        for (int cc = 0; cc < NCLASS; ++cc) l += __expf(sOut[r][cc] - m);
        float lg = m + __logf(l);
#pragma unroll
        for (int cc = 0; cc < NCLASS; ++cc)
            out[(size_t)(rowTile*16 + r)*NCLASS + cc] = sOut[r][cc] - lg;
    }
}

// ---------------------------------------------------------------------------
extern "C" void kernel_launch(void* const* d_in, const int* in_sizes, int n_in,
                              void* d_out, int out_size, void* d_ws, size_t ws_size,
                              hipStream_t stream) {
    const float* x       = (const float*)d_in[0];   // [4096,1024]
    const int*   adj     = (const int*)  d_in[1];   // [4096,4096]
    const float* W_heads = (const float*)d_in[2];   // [8,1024,64]
    const float* b_heads = (const float*)d_in[3];   // [8,64]
    const float* a_heads = (const float*)d_in[4];   // [8,128]
    const float* W_out   = (const float*)d_in[5];   // [512,32]
    const float* b_out   = (const float*)d_in[6];   // [32]
    const float* a_out   = (const float*)d_in[7];   // [64]
    float* out = (float*)d_out;                     // [4096,32]

    // workspace carve-up (~18.5 MB)
    char* p = (char*)d_ws;
    auto alloc = [&](size_t bytes) -> void* {
        void* r = (void*)p; p += (bytes + 255) & ~(size_t)255; return r;
    };
    __bf16* xb    = (__bf16*)alloc((size_t)NN*NF*2);          // x bf16
    __bf16* WtT   = (__bf16*)alloc((size_t)NHEAD*NHID*NF*2);  // W_heads^T
    __bf16* hbT   = (__bf16*)alloc((size_t)NHEAD*NHID*NN*2);  // h^T per head
    float*  ssrc  = (float*) alloc((size_t)NHEAD*NN*4);
    float*  sdst  = (float*) alloc((size_t)NHEAD*NN*4);
    float*  gmax  = (float*) alloc(256);
    __bf16* xcat  = (__bf16*)alloc((size_t)NN*NCAT*2);        // concat feats
    __bf16* WoutT = (__bf16*)alloc((size_t)NCLASS*NCAT*2);
    __bf16* houtT = (__bf16*)alloc((size_t)NCLASS*NN*2);
    float*  s2src = (float*) alloc((size_t)NN*4);
    float*  s2dst = (float*) alloc((size_t)NN*4);
    float*  gmax2 = (float*) alloc(256);

    // 1) precision casts / transposes
    k_cast_x<<<(NN*NF + 255)/256, 256, 0, stream>>>(x, xb, NN*NF);
    k_transpose_Wheads<<<dim3(NHID, NHEAD), 256, 0, stream>>>(W_heads, WtT);
    k_transpose_Wout<<<NCLASS, 256, 0, stream>>>(W_out, WoutT);

    // 2) per-head feature GEMM (WMMA bf16, async-LDS staged B) -> hbT
    k_head_gemm<<<dim3(NN/128, NHEAD), 256, 0, stream>>>(xb, WtT, b_heads, hbT);

    // 3) attention scores + global dst-max per head
    k_scores_head<<<NHEAD, 256, 0, stream>>>(hbT, a_heads, ssrc, sdst, gmax);

    // 4) fused flash attention for all heads (adj reuse via 192MB L2)
    k_attention_heads<<<dim3(NN/16, NHEAD), 128, 0, stream>>>(adj, hbT, ssrc, sdst, gmax, xcat);

    // 5) output GEMM (WMMA bf16) -> houtT
    k_out_gemm<<<NN/128, 256, 0, stream>>>(xcat, WoutT, b_out, houtT);

    // 6) output-layer scores + dst-max
    k_scores_out<<<1, 256, 0, stream>>>(houtT, a_out, s2src, s2dst, gmax2);

    // 7) fused output attention + log_softmax
    k_attention_out<<<NN/16, 128, 0, stream>>>(adj, houtT, s2src, s2dst, gmax2, out);

    (void)in_sizes; (void)n_in; (void)out_size; (void)ws_size;
}
